// INLMixtureOfExperts_3599182594274
// MI455X (gfx1250) — matrix operand order repaired and checked by reading, not compiled, not run on hardware
//
#include <hip/hip_runtime.h>

#define USE_ASYNC_COPY 1

typedef __bf16 v16bf __attribute__((ext_vector_type(16)));
typedef float  v8f   __attribute__((ext_vector_type(8)));
typedef unsigned short ushortv4 __attribute__((ext_vector_type(4)));

#define C_B    4096
#define C_D    1024
#define C_E    8
#define C_HID  512
#define C_RH   256
#define C_EMB  32

#define BM 128
#define BN 128
#define BKK 32

__device__ __forceinline__ unsigned short f2bf(float f) {
    unsigned u = __float_as_uint(f);
    unsigned r = u + 0x7FFFu + ((u >> 16) & 1u);
    return (unsigned short)(r >> 16);
}

__device__ __forceinline__ void async_copy_b128(void* ldsDst, const void* gsrc) {
#if USE_ASYNC_COPY
    unsigned l = (unsigned)(unsigned long long)ldsDst;   // low 32 bits = LDS offset
    asm volatile("global_load_async_to_lds_b128 %0, %1, off"
                 :: "v"(l), "v"(gsrc) : "memory");
#else
    *(uint4*)ldsDst = *(const uint4*)gsrc;
#endif
}
__device__ __forceinline__ void async_wait() {
#if USE_ASYNC_COPY
    asm volatile("s_wait_asynccnt 0x0" ::: "memory");
#endif
}

// ---------------------------------------------------------------------------
// Tiled bf16 WMMA GEMM. Block: 256 thr = 8 waves, tile 128x128, K-step 32,
// double-buffered LDS (async global->LDS for bf16 A, fp32->bf16 convert for W).
// Wave (4x2 grid) computes 32x64 -> 2x4 accumulators of v_wmma_f32_16x16x32_bf16.
// EPI 0: router  (A = [h|x] fp32->bf16, out fp32 += ebias)
// EPI 1: expert GEMM1 (A = Xg bf16 async, out fp32 + b1)       per-expert rows
// EPI 2: expert GEMM2 (A = Hid bf16 async, out bf16 + b2)
// EPI 3: expert head  (A = Feats slice, act(.)*wgt atomic-> d_out)
// ---------------------------------------------------------------------------
template <int EPI>
__global__ __launch_bounds__(256) void gemm_bf16(
    const float* __restrict__ hsrc, const float* __restrict__ xsrc,
    const unsigned short* __restrict__ Abf, int lda,
    const float* __restrict__ Wsrc, int Ksize, int Nsize,
    const float* __restrict__ bias,
    float* __restrict__ outF,
    unsigned short* __restrict__ outBf,
    float* __restrict__ dout, int oIdx,
    const int* __restrict__ cnt, const int* __restrict__ offs,
    const int* __restrict__ tokArr, const float* __restrict__ wgtArr)
{
    const int tid = threadIdx.x;
    const int e   = (EPI == 0) ? 0 : (int)blockIdx.z;
    const int M   = (EPI == 0) ? C_B : cnt[e];
    const int m0  = blockIdx.y * BM;
    if (EPI != 0 && m0 >= M) return;
    const int rowOff = (EPI == 0) ? 0 : offs[e];
    const int n0  = blockIdx.x * BN;
    const float* W = Wsrc + (size_t)e * (size_t)Ksize * (size_t)Nsize;

    __shared__ unsigned short As [2][BM][BKK];   // [buf][row][k]
    __shared__ unsigned short BsT[2][BN][BKK];   // [buf][col][k] (transposed)

    const int lane = tid & 31, l16 = lane & 15, lhi = lane >> 4;
    const int wave = tid >> 5;
    const int waveM = (wave & 3) * 32;
    const int waveN = (wave >> 2) * 64;

    auto loadTiles = [&](int k0, int buf) {
        // ---- A tile ----
        if (EPI == 0) {
            const float* base = (k0 < C_D) ? (hsrc + k0) : (xsrc + (k0 - C_D));
#pragma unroll
            for (int i = 0; i < 4; ++i) {
                int f = tid + i * 256;          // 1024 float4 slots
                int r = f >> 3, c4 = f & 7;
                int row = m0 + r;
                float4 v = *(const float4*)(base + (size_t)row * C_D + c4 * 4);
                ushortv4 pv;
                pv.x = f2bf(v.x); pv.y = f2bf(v.y); pv.z = f2bf(v.z); pv.w = f2bf(v.w);
                *(ushortv4*)&As[buf][r][c4 * 4] = pv;
            }
        } else {
#pragma unroll
            for (int i = 0; i < 2; ++i) {
                int f = tid + i * 256;          // 512 b128 slots
                int r = f >> 2, q = f & 3;
                int rl = m0 + r; if (rl >= M) rl = M - 1;   // clamp; masked at store
                const unsigned short* g =
                    Abf + (size_t)(rowOff + rl) * lda + k0 + q * 8;
                async_copy_b128(&As[buf][r][q * 8], g);
            }
        }
        // ---- B tile: fp32 weights -> bf16, transposed ----
#pragma unroll
        for (int i = 0; i < 4; ++i) {
            int f = tid + i * 256;              // 1024 float4 slots
            int k = f >> 5, nq = f & 31;
            float4 v = *(const float4*)(W + (size_t)(k0 + k) * Nsize + n0 + nq * 4);
            BsT[buf][nq * 4 + 0][k] = f2bf(v.x);
            BsT[buf][nq * 4 + 1][k] = f2bf(v.y);
            BsT[buf][nq * 4 + 2][k] = f2bf(v.z);
            BsT[buf][nq * 4 + 3][k] = f2bf(v.w);
        }
    };

    v8f acc[2][4];
#pragma unroll
    for (int mi = 0; mi < 2; ++mi)
#pragma unroll
        for (int ni = 0; ni < 4; ++ni)
#pragma unroll
            for (int r = 0; r < 8; ++r) acc[mi][ni][r] = 0.f;

    // prologue: fill buffer 0
    loadTiles(0, 0);
    async_wait();
    __syncthreads();

    const int nk = Ksize / BKK;
    for (int kt = 0; kt < nk; ++kt) {
        const int cur = kt & 1, nxt = cur ^ 1;
        // prefetch next tile into the other buffer (overlaps with WMMA below)
        if (kt + 1 < nk) loadTiles((kt + 1) * BKK, nxt);

        // ---- fragments (ISA 16-bit layouts: A 16x32, B 32x16) ----
        v16bf afr[2], bfr[4];
#pragma unroll
        for (int mi = 0; mi < 2; ++mi) {
            int m = waveM + mi * 16 + l16;
            ((uint4*)&afr[mi])[0] = *(const uint4*)&As[cur][m][lhi * 8];
            ((uint4*)&afr[mi])[1] = *(const uint4*)&As[cur][m][16 + lhi * 8];
        }
#pragma unroll
        for (int ni = 0; ni < 4; ++ni) {
            int n = waveN + ni * 16 + l16;
            ((uint4*)&bfr[ni])[0] = *(const uint4*)&BsT[cur][n][lhi * 8];
            ((uint4*)&bfr[ni])[1] = *(const uint4*)&BsT[cur][n][16 + lhi * 8];
        }
#pragma unroll
        for (int mi = 0; mi < 2; ++mi)
#pragma unroll
            for (int ni = 0; ni < 4; ++ni)
                acc[mi][ni] = __builtin_amdgcn_wmma_f32_16x16x32_bf16(
                    false, afr[mi], false, bfr[ni], (short)0, acc[mi][ni],
                    false, false);

        // own async writes (next buffer) complete, then workgroup barrier:
        // next buffer globally visible, current buffer fully consumed.
        async_wait();
        __syncthreads();
    }

    // ---- epilogue (C layout: vgpr r -> M = r + 8*(lane>=16), lane%16 -> N) ----
#pragma unroll
    for (int mi = 0; mi < 2; ++mi)
#pragma unroll
        for (int ni = 0; ni < 4; ++ni)
#pragma unroll
            for (int r = 0; r < 8; ++r) {
                int row = m0 + waveM + mi * 16 + lhi * 8 + r;
                int col = n0 + waveN + ni * 16 + l16;
                float v = acc[mi][ni][r];
                if (EPI == 0) {
                    outF[(size_t)row * Nsize + col] = v + bias[col];
                } else if (row < M) {
                    size_t grow = (size_t)(rowOff + row);
                    if (EPI == 1) {
                        outF[grow * Nsize + col] = v + bias[(size_t)e * Nsize + col];
                    } else if (EPI == 2) {
                        outBf[grow * (size_t)Nsize + col] =
                            f2bf(v + bias[(size_t)e * Nsize + col]);
                    } else {
                        int   token = tokArr[e * C_B + row];
                        float w     = wgtArr[e * C_B + row];
                        float z = v + bias[(size_t)e * Nsize + col];
                        float rz;
                        if (oIdx == 1)      rz = (z > 20.f) ? z : log1pf(expf(z));
                        else if (oIdx == 3) rz = z;
                        else                rz = 1.f / (1.f + expf(-z));
                        unsafeAtomicAdd(
                            &dout[((size_t)oIdx * C_B + token) * C_D + col], w * rz);
                    }
                }
            }
}

// ---------------------------------------------------------------------------
// small kernels
// ---------------------------------------------------------------------------
__global__ void zero_init(float* out, size_t n, int* cnt) {
    size_t i = (size_t)blockIdx.x * blockDim.x + threadIdx.x;
    size_t stride = (size_t)gridDim.x * blockDim.x;
    for (; i < n; i += stride) out[i] = 0.f;
    if (blockIdx.x == 0 && threadIdx.x < C_E) cnt[threadIdx.x] = 0;
}

// ebias[n] = br1[n] + layer_emb[layer_idx,:]·Wr1[2048+j, n] + phase_emb[0,:]·Wr1[2080+j, n]
__global__ void ebias_kernel(const float* __restrict__ Wr1, const float* __restrict__ br1,
                             const float* __restrict__ layer_emb,
                             const float* __restrict__ phase_emb,
                             const int* __restrict__ layer_idx,
                             float* __restrict__ ebias) {
    int n = threadIdx.x;
    int li = layer_idx[0];
    float acc = br1[n];
    for (int j = 0; j < C_EMB; ++j)
        acc += layer_emb[li * C_EMB + j] * Wr1[(size_t)(2 * C_D + j) * C_RH + n];
    for (int j = 0; j < C_EMB; ++j)
        acc += phase_emb[j] * Wr1[(size_t)(2 * C_D + C_EMB + j) * C_RH + n];
    ebias[n] = acc;
}

// LN + exact gelu + 256x8 logits + top2 + softmax -> per-expert lists
__global__ __launch_bounds__(256) void router_finish(
    const float* __restrict__ rhid_pre,
    const float* __restrict__ lnr_g, const float* __restrict__ lnr_b,
    const float* __restrict__ Wr2, const float* __restrict__ br2,
    int* __restrict__ cnt, int* __restrict__ tokArr, float* __restrict__ wgtArr)
{
    int b = blockIdx.x, tid = threadIdx.x;
    float y = rhid_pre[(size_t)b * C_RH + tid];
    float ss = y, sq = y * y;
    for (int m = 16; m; m >>= 1) { ss += __shfl_xor(ss, m); sq += __shfl_xor(sq, m); }
    __shared__ float rs[8], rq[8], s[C_RH], slog[C_E];
    __shared__ float smu, srv;
    if ((tid & 31) == 0) { rs[tid >> 5] = ss; rq[tid >> 5] = sq; }
    __syncthreads();
    if (tid == 0) {
        float S = 0.f, Q = 0.f;
        for (int i = 0; i < 8; ++i) { S += rs[i]; Q += rq[i]; }
        float mu = S / 256.f;
        smu = mu; srv = rsqrtf(Q / 256.f - mu * mu + 1e-5f);
    }
    __syncthreads();
    float z = (y - smu) * srv * lnr_g[tid] + lnr_b[tid];
    s[tid] = 0.5f * z * (1.f + erff(z * 0.70710678118654752f));
    __syncthreads();
    if (tid < C_E) {
        float acc = br2[tid];
        for (int j = 0; j < C_RH; ++j) acc += s[j] * Wr2[j * C_E + tid];
        slog[tid] = acc;
    }
    __syncthreads();
    if (tid == 0) {
        int i1 = 0; float v1 = slog[0];
        for (int e = 1; e < C_E; ++e) if (slog[e] > v1) { v1 = slog[e]; i1 = e; }
        int i2 = (i1 == 0) ? 1 : 0; float v2 = slog[i2];
        for (int e = 0; e < C_E; ++e)
            if (e != i1 && slog[e] > v2) { v2 = slog[e]; i2 = e; }
        float w1 = 1.f / (1.f + expf(v2 - v1));
        float w2 = 1.f - w1;
        int s1 = atomicAdd(&cnt[i1], 1);
        tokArr[i1 * C_B + s1] = b; wgtArr[i1 * C_B + s1] = w1;
        int s2 = atomicAdd(&cnt[i2], 1);
        tokArr[i2 * C_B + s2] = b; wgtArr[i2 * C_B + s2] = w2;
    }
}

__global__ void offsets_kernel(const int* __restrict__ cnt, int* __restrict__ offs) {
    if (threadIdx.x == 0) {
        int a = 0;
        for (int e = 0; e < C_E; ++e) { offs[e] = a; a += cnt[e]; }
    }
}

// gather compacted hx rows -> Xg bf16 [8192 x 2048]
__global__ __launch_bounds__(256) void gather_hx(
    const float* __restrict__ h, const float* __restrict__ x,
    const int* __restrict__ cnt, const int* __restrict__ offs,
    const int* __restrict__ tokArr, unsigned short* __restrict__ Xg)
{
    int e = blockIdx.y;
    int s0 = blockIdx.x * 8;
    int nE = cnt[e];
    if (s0 >= nE) return;
    int sl = s0 + (threadIdx.x >> 5);
    if (sl >= nE) return;
    int lane = threadIdx.x & 31;
    int token = tokArr[e * C_B + sl];
    size_t g = (size_t)(offs[e] + sl);
    for (int i = 0; i < 16; ++i) {
        int c4 = lane + i * 32;          // 512 float4 per row
        int c  = c4 * 4;
        const float* src = (c < C_D) ? (h + (size_t)token * C_D + c)
                                     : (x + (size_t)token * C_D + c - C_D);
        float4 v = *(const float4*)src;
        ushortv4 pv;
        pv.x = f2bf(v.x); pv.y = f2bf(v.y); pv.z = f2bf(v.z); pv.w = f2bf(v.w);
        *(ushortv4*)&Xg[g * (2 * C_D) + c] = pv;
    }
}

// row-wise LN(512) + exact gelu -> Hid bf16
__global__ __launch_bounds__(256) void ln_gelu_hid(
    const float* __restrict__ Hpre, const float* __restrict__ g1,
    const float* __restrict__ b1, const int* __restrict__ offs,
    unsigned short* __restrict__ Hid)
{
    int grow = blockIdx.x;
    int e = 0;
    while (e < C_E - 1 && grow >= offs[e + 1]) ++e;
    int tid = threadIdx.x;
    float y0 = Hpre[(size_t)grow * C_HID + tid];
    float y1 = Hpre[(size_t)grow * C_HID + tid + 256];
    float ss = y0 + y1, sq = y0 * y0 + y1 * y1;
    for (int m = 16; m; m >>= 1) { ss += __shfl_xor(ss, m); sq += __shfl_xor(sq, m); }
    __shared__ float rs[8], rq[8];
    __shared__ float smu, srv;
    if ((tid & 31) == 0) { rs[tid >> 5] = ss; rq[tid >> 5] = sq; }
    __syncthreads();
    if (tid == 0) {
        float S = 0.f, Q = 0.f;
        for (int i = 0; i < 8; ++i) { S += rs[i]; Q += rq[i]; }
        float mu = S / 512.f;
        smu = mu; srv = rsqrtf(Q / 512.f - mu * mu + 1e-5f);
    }
    __syncthreads();
    float mu = smu, rv = srv;
    int c0 = tid, c1 = tid + 256;
    float z0 = (y0 - mu) * rv * g1[e * C_HID + c0] + b1[e * C_HID + c0];
    float z1 = (y1 - mu) * rv * g1[e * C_HID + c1] + b1[e * C_HID + c1];
    z0 = 0.5f * z0 * (1.f + erff(z0 * 0.70710678118654752f));
    z1 = 0.5f * z1 * (1.f + erff(z1 * 0.70710678118654752f));
    Hid[(size_t)grow * C_HID + c0] = f2bf(z0);
    Hid[(size_t)grow * C_HID + c1] = f2bf(z1);
}

// ---------------------------------------------------------------------------
extern "C" void kernel_launch(void* const* d_in, const int* in_sizes, int n_in,
                              void* d_out, int out_size, void* d_ws, size_t ws_size,
                              hipStream_t stream) {
    const float* h         = (const float*)d_in[0];
    const float* x         = (const float*)d_in[1];
    const int*   layer_idx = (const int*)d_in[2];
    const float* layer_emb = (const float*)d_in[3];
    const float* phase_emb = (const float*)d_in[4];
    const float* Wr1       = (const float*)d_in[5];
    const float* br1       = (const float*)d_in[6];
    const float* lnr_g     = (const float*)d_in[7];
    const float* lnr_b     = (const float*)d_in[8];
    const float* Wr2       = (const float*)d_in[9];
    const float* br2       = (const float*)d_in[10];
    const float* W1        = (const float*)d_in[11];
    const float* b1        = (const float*)d_in[12];
    const float* ln1_g     = (const float*)d_in[13];
    const float* ln1_b     = (const float*)d_in[14];
    const float* W2        = (const float*)d_in[15];
    const float* b2        = (const float*)d_in[16];
    const float* Wa        = (const float*)d_in[17];
    const float* ba        = (const float*)d_in[18];
    const float* Wb        = (const float*)d_in[19];
    const float* bb        = (const float*)d_in[20];
    const float* Wg        = (const float*)d_in[21];
    const float* bg        = (const float*)d_in[22];
    const float* Wv        = (const float*)d_in[23];
    const float* bv        = (const float*)d_in[24];
    float* out = (float*)d_out;

    const int TOT = C_B * 2;                       // 8192 compacted rows
    char* w = (char*)d_ws;
    float* ebias = (float*)w;            w += 1024;
    int*   cnt   = (int*)w;              w += 256;
    int*   offs  = (int*)w;              w += 256;
    int*   tokA  = (int*)w;              w += (size_t)C_E * C_B * 4;
    float* wgtA  = (float*)w;            w += (size_t)C_E * C_B * 4;
    float* rhid  = (float*)w;            w += (size_t)C_B * C_RH * 4;
    unsigned short* Xg    = (unsigned short*)w; w += (size_t)TOT * 2 * C_D * 2;
    float*          Hpre  = (float*)w;          w += (size_t)TOT * C_HID * 4;
    unsigned short* Hid   = (unsigned short*)w; w += (size_t)TOT * C_HID * 2;
    unsigned short* Feats = (unsigned short*)w; w += (size_t)TOT * 4 * C_D * 2;

    // 0) zero out + counters
    zero_init<<<4096, 256, 0, stream>>>(out, (size_t)4 * C_B * C_D, cnt);
    // 1) fold embeddings + br1 into router bias
    ebias_kernel<<<1, 256, 0, stream>>>(Wr1, br1, layer_emb, phase_emb, layer_idx, ebias);
    // 2) router GEMM: [h|x](4096x2048) @ Wr1(2048x256) -> rhid_pre
    gemm_bf16<0><<<dim3(C_RH / BN, C_B / BM, 1), 256, 0, stream>>>(
        h, x, nullptr, 0, Wr1, 2 * C_D, C_RH, ebias, rhid, nullptr,
        nullptr, 0, nullptr, nullptr, nullptr, nullptr);
    // 3) LN+gelu+logits+top2+softmax -> per-expert lists
    router_finish<<<C_B, 256, 0, stream>>>(rhid, lnr_g, lnr_b, Wr2, br2, cnt, tokA, wgtA);
    offsets_kernel<<<1, 32, 0, stream>>>(cnt, offs);
    // 4) gather compacted bf16 activations
    gather_hx<<<dim3(C_B / 8, C_E, 1), 256, 0, stream>>>(h, x, cnt, offs, tokA, Xg);
    // 5) GEMM1: Xg(8192x2048) @ W1_e(2048x512) -> Hpre
    gemm_bf16<1><<<dim3(C_HID / BN, C_B / BM, C_E), 256, 0, stream>>>(
        nullptr, nullptr, Xg, 2 * C_D, W1, 2 * C_D, C_HID, b1, Hpre, nullptr,
        nullptr, 0, cnt, offs, tokA, wgtA);
    // 6) LN + gelu -> Hid bf16
    ln_gelu_hid<<<TOT, 256, 0, stream>>>(Hpre, ln1_g, ln1_b, offs, Hid);
    // 7) GEMM2: Hid(8192x512) @ W2_e(512x4096) -> Feats bf16
    gemm_bf16<2><<<dim3(4 * C_D / BN, C_B / BM, C_E), 256, 0, stream>>>(
        nullptr, nullptr, Hid, C_HID, W2, C_HID, 4 * C_D, b2, nullptr, Feats,
        nullptr, 0, cnt, offs, tokA, wgtA);
    // 8) four head GEMMs: act(F_o @ W_o + b_o) * wgt -> atomic accumulate d_out
    const float* Wh[4] = { Wa, Wb, Wg, Wv };
    const float* bh[4] = { ba, bb, bg, bv };
    for (int o = 0; o < 4; ++o) {
        gemm_bf16<3><<<dim3(C_D / BN, C_B / BM, C_E), 256, 0, stream>>>(
            nullptr, nullptr, Feats + o * C_D, 4 * C_D, Wh[o], C_D, C_D, bh[o],
            nullptr, nullptr, out, o, cnt, offs, tokA, wgtA);
    }
}
